// FlexScaleSeqAttentionV1_36051955482843
// MI455X (gfx1250) — matrix operand order repaired and checked by reading, not compiled, not run on hardware
//
#include <hip/hip_runtime.h>

// ---------------------------------------------------------------------------
// FlexScaleSeqAttention for MI455X (gfx1250), bf16 WMMA (16x16x32, f32 acc).
// Refactored: out = w_oseq @ (attn^T @ v) + b_oseq * colsum(v)
// All GEMM B-operands staged to LDS as [N][K] via contiguous b128 copies
// (x pre-transposed during cast; attn^T@v result stored transposed).
// ---------------------------------------------------------------------------

typedef __attribute__((ext_vector_type(16))) __bf16 v16bf;
typedef __attribute__((ext_vector_type(8)))  __bf16 bf16x8;
typedef __attribute__((ext_vector_type(8)))  float  v8f;

#define NB   2
#define SEQ  2048
#define CIN  1024
#define NH   16
#define HD   64

static __device__ inline v8f zero8() {
  v8f z = {0.f, 0.f, 0.f, 0.f, 0.f, 0.f, 0.f, 0.f};
  return z;
}

static __device__ inline v8f wmma_bf(v16bf a, v16bf b, v8f c) {
  return __builtin_amdgcn_wmma_f32_16x16x32_bf16(false, a, false, b,
                                                 (short)0, c, false, false);
}

// A-matrix fragment (16x32 bf16): lane = row, halves interleave K in 8-elem
// groups: elems 0..7 -> K = half*8 + i ; elems 8..15 -> K = 16 + half*8 + i
static __device__ inline v16bf frag_a(const __bf16* base, int stride) {
  int lane = threadIdx.x & 31;
  const __bf16* p = base + (lane & 15) * stride + (lane >> 4) * 8;
  bf16x8 g0 = *(const bf16x8*)p;
  bf16x8 g1 = *(const bf16x8*)(p + 16);
  v16bf f;
#pragma unroll
  for (int i = 0; i < 8; ++i) { f[i] = g0[i]; f[i + 8] = g1[i]; }
  return f;
}

// B-matrix fragment (32x16 bf16), LDS stored [N][K]: lane = col n, halves
// interleave K: elems 0..7 -> K = half*8 + i ; elems 8..15 -> K = 16 + half*8 + i
static __device__ inline v16bf frag_b(const __bf16* base, int stride) {
  int lane = threadIdx.x & 31;
  const __bf16* p = base + (lane & 15) * stride + (lane >> 4) * 8;
  bf16x8 g0 = *(const bf16x8*)p;
  bf16x8 g1 = *(const bf16x8*)(p + 16);
  v16bf f;
#pragma unroll
  for (int i = 0; i < 8; ++i) { f[i] = g0[i]; f[i + 8] = g1[i]; }
  return f;
}

// Stage 128 rows x 32 cols of bf16 [row-major, ld] into LDS [128][40]
static __device__ inline void stage128x32(__bf16* dst, const __bf16* src,
                                          int ld, int t) {
  int r = t >> 1, h2 = t & 1;
  const __bf16* g = src + (size_t)r * ld + h2 * 16;
  *(bf16x8*)&dst[r * 40 + h2 * 16]     = *(const bf16x8*)g;
  *(bf16x8*)&dst[r * 40 + h2 * 16 + 8] = *(const bf16x8*)(g + 8);
}

// ---------------------------------------------------------------------------
__global__ void __launch_bounds__(256) k_cast(const float* __restrict__ in,
                                              __bf16* __restrict__ out, int n) {
  int i = blockIdx.x * 256 + threadIdx.x;
  int stride = gridDim.x * 256;
  for (; i < n; i += stride) out[i] = (__bf16)in[i];
}

// ---------------------------------------------------------------------------
// x [B, SEQ, CIN] f32  ->  xt [B, CIN, SEQ] bf16 (32x32 LDS tile transpose)
__global__ void __launch_bounds__(256) k_xpose(const float* __restrict__ x,
                                               __bf16* __restrict__ xt) {
  int b = blockIdx.z, s0 = blockIdx.x * 32, c0 = blockIdx.y * 32;
  __shared__ float tile[32][33];
  int t = threadIdx.x, tc = t & 31, tr = t >> 5;  // 8 rows per pass
  const float* src = x + ((size_t)b * SEQ + s0) * CIN + c0;
#pragma unroll
  for (int i = 0; i < 4; ++i)
    tile[tr + i * 8][tc] = src[(size_t)(tr + i * 8) * CIN + tc];
  __syncthreads();
  int ts = t & 31, cc = t >> 5;
  __bf16* dst = xt + ((size_t)b * CIN + c0) * SEQ + s0;
#pragma unroll
  for (int i = 0; i < 4; ++i)
    dst[(size_t)(cc + i * 8) * SEQ + ts] = (__bf16)tile[ts][cc + i * 8];
}

// ---------------------------------------------------------------------------
// xh[b,t,c] = sum_s w_hseq[t,s] * xt[b,c,s] + b_hseq[t]
// Block tile 128(M) x 128(N); 8 waves, each 32x64; BK=32.
__global__ void __launch_bounds__(256) k_gemm_seq(
    const __bf16* __restrict__ A, const __bf16* __restrict__ Xt,
    const float* __restrict__ bias_row, __bf16* __restrict__ Out) {
  int b = blockIdx.z;
  int m0 = blockIdx.y * 128;
  int n0 = blockIdx.x * 128;
  const __bf16* Bbase = Xt + (size_t)b * CIN * SEQ;
  __bf16* obase = Out + (size_t)b * SEQ * CIN;
  __shared__ alignas(16) __bf16 Al[128 * 40];
  __shared__ alignas(16) __bf16 Bl[128 * 40];
  int t = threadIdx.x, w = t >> 5, lane = t & 31, ln = lane & 15, hf = lane >> 4;
  int wm = (w >> 1) * 32, wn = (w & 1) * 64;
  v8f acc[2][4];
#pragma unroll
  for (int i = 0; i < 2; ++i)
#pragma unroll
    for (int j = 0; j < 4; ++j) acc[i][j] = zero8();

  for (int kb = 0; kb < SEQ; kb += 32) {
    __syncthreads();
    stage128x32(Al, A + (size_t)m0 * SEQ + kb, SEQ, t);
    stage128x32(Bl, Bbase + (size_t)n0 * SEQ + kb, SEQ, t);
    if (kb + 32 < SEQ) {
      __builtin_prefetch(A + (size_t)(m0 + (t >> 1)) * SEQ + kb + 32, 0, 1);
      __builtin_prefetch(Bbase + (size_t)(n0 + (t >> 1)) * SEQ + kb + 32, 0, 1);
    }
    __syncthreads();
    v16bf a0 = frag_a(&Al[(wm + 0) * 40], 40);
    v16bf a1 = frag_a(&Al[(wm + 16) * 40], 40);
    v16bf bf[4];
#pragma unroll
    for (int tn = 0; tn < 4; ++tn) bf[tn] = frag_b(&Bl[(wn + tn * 16) * 40], 40);
#pragma unroll
    for (int tn = 0; tn < 4; ++tn) {
      acc[0][tn] = wmma_bf(a0, bf[tn], acc[0][tn]);
      acc[1][tn] = wmma_bf(a1, bf[tn], acc[1][tn]);
    }
  }
#pragma unroll
  for (int tm = 0; tm < 2; ++tm)
#pragma unroll
    for (int tn = 0; tn < 4; ++tn)
#pragma unroll
      for (int r = 0; r < 8; ++r) {
        int row = m0 + wm + tm * 16 + hf * 8 + r;
        int col = n0 + wn + tn * 16 + ln;
        float v = acc[tm][tn][r] + bias_row[row];
        obase[(size_t)row * CIN + col] = (__bf16)v;
      }
}

// ---------------------------------------------------------------------------
// P[b,t,o] = sum_c xh[b,t,c] * W[o,c] + bias[o], scattered to [b, h, t, d]
__global__ void __launch_bounds__(256) k_gemm_qkv(
    const __bf16* __restrict__ Xh, const __bf16* __restrict__ W,
    const float* __restrict__ bias, __bf16* __restrict__ Out) {
  int b = blockIdx.z;
  int m0 = blockIdx.y * 128;
  int n0 = blockIdx.x * 128;
  const __bf16* Abase = Xh + (size_t)b * SEQ * CIN;
  __shared__ alignas(16) __bf16 Al[128 * 40];
  __shared__ alignas(16) __bf16 Bl[128 * 40];
  int t = threadIdx.x, w = t >> 5, lane = t & 31, ln = lane & 15, hf = lane >> 4;
  int wm = (w >> 1) * 32, wn = (w & 1) * 64;
  v8f acc[2][4];
#pragma unroll
  for (int i = 0; i < 2; ++i)
#pragma unroll
    for (int j = 0; j < 4; ++j) acc[i][j] = zero8();

  for (int kb = 0; kb < CIN; kb += 32) {
    __syncthreads();
    stage128x32(Al, Abase + (size_t)m0 * CIN + kb, CIN, t);
    stage128x32(Bl, W + (size_t)n0 * CIN + kb, CIN, t);
    __syncthreads();
    v16bf a0 = frag_a(&Al[(wm + 0) * 40], 40);
    v16bf a1 = frag_a(&Al[(wm + 16) * 40], 40);
    v16bf bf[4];
#pragma unroll
    for (int tn = 0; tn < 4; ++tn) bf[tn] = frag_b(&Bl[(wn + tn * 16) * 40], 40);
#pragma unroll
    for (int tn = 0; tn < 4; ++tn) {
      acc[0][tn] = wmma_bf(a0, bf[tn], acc[0][tn]);
      acc[1][tn] = wmma_bf(a1, bf[tn], acc[1][tn]);
    }
  }
#pragma unroll
  for (int tm = 0; tm < 2; ++tm)
#pragma unroll
    for (int tn = 0; tn < 4; ++tn)
#pragma unroll
      for (int r = 0; r < 8; ++r) {
        int s = m0 + wm + tm * 16 + hf * 8 + r;
        int o = n0 + wn + tn * 16 + ln;
        int h = o >> 6, d = o & 63;
        float v = acc[tm][tn][r] + bias[o];
        Out[(((size_t)b * NH + h) * SEQ + s) * HD + d] = (__bf16)v;
      }
}

// ---------------------------------------------------------------------------
// Pass 1: per-query softmax stats m[q], l[q] over s[q,k] = 0.125 * q.k
__global__ void __launch_bounds__(256) k_stats(
    const __bf16* __restrict__ Qm, const __bf16* __restrict__ Km,
    float* __restrict__ mst, float* __restrict__ lst) {
  int bh = blockIdx.z * NH + blockIdx.y;
  int q0 = blockIdx.x * 128;
  const __bf16* Q = Qm + (size_t)bh * SEQ * HD;
  const __bf16* K = Km + (size_t)bh * SEQ * HD;
  __shared__ alignas(16) __bf16 Ql[128 * 72];
  __shared__ alignas(16) __bf16 Kl[128 * 72];
  int t = threadIdx.x, w = t >> 5, lane = t & 31, ln = lane & 15, hf = lane >> 4;
  {
    int r = t >> 1, h2 = (t & 1) * 32;
    const __bf16* src = Q + (size_t)(q0 + r) * HD + h2;
#pragma unroll
    for (int j = 0; j < 4; ++j)
      *(bf16x8*)&Ql[r * 72 + h2 + j * 8] = *(const bf16x8*)(src + j * 8);
  }
  __syncthreads();
  v16bf aq0 = frag_a(&Ql[(w * 16) * 72 + 0], 72);
  v16bf aq1 = frag_a(&Ql[(w * 16) * 72 + 32], 72);
  float mr[8], lr[8];
#pragma unroll
  for (int r = 0; r < 8; ++r) { mr[r] = -1e30f; lr[r] = 0.f; }

  for (int kb = 0; kb < SEQ; kb += 128) {
    __syncthreads();
    {
      int r = t >> 1, h2 = (t & 1) * 32;
      const __bf16* src = K + (size_t)(kb + r) * HD + h2;
#pragma unroll
      for (int j = 0; j < 4; ++j)
        *(bf16x8*)&Kl[r * 72 + h2 + j * 8] = *(const bf16x8*)(src + j * 8);
    }
    __syncthreads();
#pragma unroll
    for (int nt = 0; nt < 8; ++nt) {
      v16bf b0 = frag_b(&Kl[(nt * 16) * 72 + 0], 72);
      v16bf b1 = frag_b(&Kl[(nt * 16) * 72 + 32], 72);
      v8f s = zero8();
      s = wmma_bf(aq0, b0, s);
      s = wmma_bf(aq1, b1, s);
#pragma unroll
      for (int r = 0; r < 8; ++r) {
        float v = s[r] * 0.125f;
        float tm = v;
        tm = fmaxf(tm, __shfl_xor(tm, 1, 32));
        tm = fmaxf(tm, __shfl_xor(tm, 2, 32));
        tm = fmaxf(tm, __shfl_xor(tm, 4, 32));
        tm = fmaxf(tm, __shfl_xor(tm, 8, 32));
        float mn = fmaxf(mr[r], tm);
        float p = __expf(v - mn);
        p += __shfl_xor(p, 1, 32);
        p += __shfl_xor(p, 2, 32);
        p += __shfl_xor(p, 4, 32);
        p += __shfl_xor(p, 8, 32);
        lr[r] = lr[r] * __expf(mr[r] - mn) + p;
        mr[r] = mn;
      }
    }
  }
  if (ln == 0) {
#pragma unroll
    for (int r = 0; r < 8; ++r) {
      int row = q0 + w * 16 + hf * 8 + r;
      mst[(size_t)bh * SEQ + row] = mr[r];
      lst[(size_t)bh * SEQ + row] = lr[r];
    }
  }
}

// ---------------------------------------------------------------------------
// vsum[bh,d] = sum_q v[bh,q,d]
__global__ void __launch_bounds__(256) k_vsum(const __bf16* __restrict__ Vm,
                                              float* __restrict__ vs) {
  int bh = blockIdx.x;
  int t = threadIdx.x, d = t & 63, seg = t >> 6;
  const __bf16* V = Vm + (size_t)bh * SEQ * HD;
  float s = 0.f;
  for (int q = seg; q < SEQ; q += 4) s += (float)V[(size_t)q * HD + d];
  __shared__ float red[256];
  red[t] = s;
  __syncthreads();
  if (seg == 0)
    vs[(size_t)bh * HD + d] = red[d] + red[64 + d] + red[128 + d] + red[192 + d];
}

// ---------------------------------------------------------------------------
// Pass 2: M2t[bh,d,k] = sum_q exp(0.125*q.k - m[q])/l[q] * v[q,d]  (transposed)
__global__ void __launch_bounds__(256) k_attv(
    const __bf16* __restrict__ Qm, const __bf16* __restrict__ Km,
    const __bf16* __restrict__ Vm, const float* __restrict__ mst,
    const float* __restrict__ lst, __bf16* __restrict__ M2t) {
  int bh = blockIdx.z * NH + blockIdx.y;
  int k0 = blockIdx.x * 128;
  const __bf16* Q = Qm + (size_t)bh * SEQ * HD;
  const __bf16* K = Km + (size_t)bh * SEQ * HD;
  const __bf16* V = Vm + (size_t)bh * SEQ * HD;
  __shared__ alignas(16) __bf16 Kl[128 * 72];
  __shared__ alignas(16) __bf16 Ql[32 * 72];
  __shared__ alignas(16) __bf16 Vt[64 * 40];   // V^T chunk [d][q]
  __shared__ alignas(16) __bf16 Pl[128 * 40];  // P^T tile [k][q]
  int t = threadIdx.x, w = t >> 5, lane = t & 31, ln = lane & 15, hf = lane >> 4;
  {
    int r = t >> 1, h2 = (t & 1) * 32;
    const __bf16* src = K + (size_t)(k0 + r) * HD + h2;
#pragma unroll
    for (int j = 0; j < 4; ++j)
      *(bf16x8*)&Kl[r * 72 + h2 + j * 8] = *(const bf16x8*)(src + j * 8);
  }
  __syncthreads();
  v16bf ak0 = frag_a(&Kl[(w * 16) * 72 + 0], 72);
  v16bf ak1 = frag_a(&Kl[(w * 16) * 72 + 32], 72);
  v8f accd[4];
#pragma unroll
  for (int dt = 0; dt < 4; ++dt) accd[dt] = zero8();

  for (int qc = 0; qc < SEQ; qc += 32) {
    __syncthreads();
    {
      int r = t >> 3, c0 = (t & 7) * 8;
      *(bf16x8*)&Ql[r * 72 + c0] = *(const bf16x8*)&Q[(size_t)(qc + r) * HD + c0];
    }
    {
      int d = t & 63, q8 = (t >> 6) * 8;
#pragma unroll
      for (int i = 0; i < 8; ++i)
        Vt[d * 40 + q8 + i] = V[(size_t)(qc + q8 + i) * HD + d];
    }
    __syncthreads();
#pragma unroll
    for (int nt = 0; nt < 2; ++nt) {
      v16bf bq0 = frag_b(&Ql[(nt * 16) * 72 + 0], 72);
      v16bf bq1 = frag_b(&Ql[(nt * 16) * 72 + 32], 72);
      v8f s = zero8();
      s = wmma_bf(ak0, bq0, s);
      s = wmma_bf(ak1, bq1, s);
      int qcol = qc + nt * 16 + ln;
      float mi = mst[(size_t)bh * SEQ + qcol];
      float inv = 1.f / lst[(size_t)bh * SEQ + qcol];
#pragma unroll
      for (int r = 0; r < 8; ++r) {
        float p = __expf(s[r] * 0.125f - mi) * inv;
        int krow = w * 16 + hf * 8 + r;
        Pl[krow * 40 + nt * 16 + ln] = (__bf16)p;
      }
    }
    __syncthreads();
    v16bf ap = frag_a(&Pl[(w * 16) * 40], 40);
#pragma unroll
    for (int dt = 0; dt < 4; ++dt) {
      v16bf bv = frag_b(&Vt[(dt * 16) * 40], 40);
      accd[dt] = wmma_bf(ap, bv, accd[dt]);
    }
  }
  // store transposed: M2t[bh][d][k]
#pragma unroll
  for (int dt = 0; dt < 4; ++dt)
#pragma unroll
    for (int r = 0; r < 8; ++r) {
      int krow = k0 + w * 16 + hf * 8 + r;
      int d = dt * 16 + ln;
      M2t[((size_t)bh * HD + d) * SEQ + krow] = (__bf16)accd[dt][r];
    }
}

// ---------------------------------------------------------------------------
// out[b,o,h*64+d] = sum_k w_oseq[o,k] * M2t[bh,d,k] + b_oseq[o] * vsum[bh,d]
// Block tile 128(M=o) x 64(N=d full); 8 waves, each 32x32.
__global__ void __launch_bounds__(256) k_gemm_out(
    const __bf16* __restrict__ Wo, const __bf16* __restrict__ M2t,
    const float* __restrict__ b_oseq, const float* __restrict__ vsum,
    float* __restrict__ Out) {
  int b = blockIdx.z, h = blockIdx.y, m0 = blockIdx.x * 128;
  int bh = b * NH + h;
  const __bf16* Bbase = M2t + (size_t)bh * HD * SEQ;  // [d][k]
  __shared__ alignas(16) __bf16 Al[128 * 40];
  __shared__ alignas(16) __bf16 Bl[64 * 40];
  int t = threadIdx.x, w = t >> 5, lane = t & 31, ln = lane & 15, hf = lane >> 4;
  int wm = (w >> 1) * 32, wn = (w & 1) * 32;
  v8f acc[2][2];
  acc[0][0] = zero8(); acc[0][1] = zero8();
  acc[1][0] = zero8(); acc[1][1] = zero8();

  for (int kb = 0; kb < SEQ; kb += 32) {
    __syncthreads();
    stage128x32(Al, Wo + (size_t)m0 * SEQ + kb, SEQ, t);
    {
      int r = t >> 2, q4 = t & 3;
      *(bf16x8*)&Bl[r * 40 + q4 * 8] =
          *(const bf16x8*)&Bbase[(size_t)r * SEQ + kb + q4 * 8];
    }
    if (kb + 32 < SEQ)
      __builtin_prefetch(Wo + (size_t)(m0 + (t >> 1)) * SEQ + kb + 32, 0, 1);
    __syncthreads();
    v16bf a0 = frag_a(&Al[(wm + 0) * 40], 40);
    v16bf a1 = frag_a(&Al[(wm + 16) * 40], 40);
    v16bf b0 = frag_b(&Bl[(wn + 0) * 40], 40);
    v16bf b1 = frag_b(&Bl[(wn + 16) * 40], 40);
    acc[0][0] = wmma_bf(a0, b0, acc[0][0]);
    acc[0][1] = wmma_bf(a0, b1, acc[0][1]);
    acc[1][0] = wmma_bf(a1, b0, acc[1][0]);
    acc[1][1] = wmma_bf(a1, b1, acc[1][1]);
  }
#pragma unroll
  for (int tm = 0; tm < 2; ++tm)
#pragma unroll
    for (int tn = 0; tn < 2; ++tn)
#pragma unroll
      for (int r = 0; r < 8; ++r) {
        int o = m0 + wm + tm * 16 + hf * 8 + r;
        int d = wn + tn * 16 + ln;
        float v = acc[tm][tn][r] + b_oseq[o] * vsum[(size_t)bh * HD + d];
        Out[((size_t)b * SEQ + o) * CIN + h * HD + d] = v;
      }
}

// ---------------------------------------------------------------------------
extern "C" void kernel_launch(void* const* d_in, const int* in_sizes, int n_in,
                              void* d_out, int out_size, void* d_ws,
                              size_t ws_size, hipStream_t stream) {
  (void)in_sizes; (void)n_in; (void)out_size; (void)ws_size;
  const float* x      = (const float*)d_in[0];
  const float* w_hseq = (const float*)d_in[1];
  const float* b_hseq = (const float*)d_in[2];
  const float* wq     = (const float*)d_in[3];
  const float* bq     = (const float*)d_in[4];
  const float* wk     = (const float*)d_in[5];
  const float* bk     = (const float*)d_in[6];
  const float* wv     = (const float*)d_in[7];
  const float* bv     = (const float*)d_in[8];
  const float* w_oseq = (const float*)d_in[9];
  const float* b_oseq = (const float*)d_in[10];
  float* out = (float*)d_out;

  const size_t NXC = (size_t)NB * SEQ * CIN;  // 4194304
  const size_t NWW = (size_t)SEQ * SEQ;       // 4194304
  const size_t NW1 = (size_t)CIN * CIN;       // 1048576

  __bf16* p = (__bf16*)d_ws;
  __bf16* xt   = p;  p += NXC;  // x transposed [B][CIN][SEQ]
  __bf16* whs  = p;  p += NWW;
  __bf16* wqb  = p;  p += NW1;
  __bf16* wkb  = p;  p += NW1;
  __bf16* wvb  = p;  p += NW1;
  __bf16* wob  = p;  p += NWW;
  __bf16* xh   = p;  p += NXC;
  __bf16* qb   = p;  p += NXC;
  __bf16* kb   = p;  p += NXC;
  __bf16* vb   = p;  p += NXC;
  __bf16* m2t  = p;  p += NXC;  // [B*H][HD][SEQ]
  float* mst = (float*)p;
  float* lst = mst + (size_t)NB * NH * SEQ;
  float* vs  = lst + (size_t)NB * NH * SEQ;

  dim3 blk(256);
  k_xpose<<<dim3(SEQ / 32, CIN / 32, NB), blk, 0, stream>>>(x, xt);
  k_cast<<<4096, blk, 0, stream>>>(w_hseq, whs, (int)NWW);
  k_cast<<<1024, blk, 0, stream>>>(wq,     wqb, (int)NW1);
  k_cast<<<1024, blk, 0, stream>>>(wk,     wkb, (int)NW1);
  k_cast<<<1024, blk, 0, stream>>>(wv,     wvb, (int)NW1);
  k_cast<<<4096, blk, 0, stream>>>(w_oseq, wob, (int)NWW);

  k_gemm_seq<<<dim3(CIN / 128, SEQ / 128, NB), blk, 0, stream>>>(whs, xt, b_hseq, xh);
  k_gemm_qkv<<<dim3(CIN / 128, SEQ / 128, NB), blk, 0, stream>>>(xh, wqb, bq, qb);
  k_gemm_qkv<<<dim3(CIN / 128, SEQ / 128, NB), blk, 0, stream>>>(xh, wkb, bk, kb);
  k_gemm_qkv<<<dim3(CIN / 128, SEQ / 128, NB), blk, 0, stream>>>(xh, wvb, bv, vb);
  k_stats<<<dim3(SEQ / 128, NH, NB), blk, 0, stream>>>(qb, kb, mst, lst);
  k_vsum<<<NB * NH, blk, 0, stream>>>(vb, vs);
  k_attv<<<dim3(SEQ / 128, NH, NB), blk, 0, stream>>>(qb, kb, vb, mst, lst, m2t);
  k_gemm_out<<<dim3(SEQ / 128, NH, NB), blk, 0, stream>>>(wob, m2t, b_oseq, vs, out);
}
